// MultiHeadAttention_25469156065624
// MI455X (gfx1250) — matrix operand order repaired and checked by reading, not compiled, not run on hardware
//
#include <hip/hip_runtime.h>
#include <hip/hip_bf16.h>

typedef __attribute__((ext_vector_type(16))) _Float16 v16h;
typedef __attribute__((ext_vector_type(8)))  _Float16 v8h;
typedef __attribute__((ext_vector_type(8)))  float    v8f;

#define S_LEN 2048
#define BATCH 2
#define DMODEL 1024
#define NHEAD 16
#define HDIM 64
#define MROWS (S_LEN * BATCH)   // 4096

// ---------------------------------------------------------------------------
// WMMA helpers (v_wmma_f32_16x16x32_f16; layouts per CDNA5 ISA)
// ---------------------------------------------------------------------------
__device__ __forceinline__ v8f wmma_f16(v16h a, v16h b, v8f c) {
  return __builtin_amdgcn_wmma_f32_16x16x32_f16(
      /*neg_a=*/false, a, /*neg_b=*/false, b,
      /*c_mod=*/(short)0, c, /*reuse_a=*/false, /*reuse_b=*/false);
}

// A-fragment: 16x32 f16, src row-major [m][k], leading dim ld (halves).
__device__ __forceinline__ v16h load_a_frag(const _Float16* base, int ld,
                                            int m0, int k0) {
  int lane = threadIdx.x & 31;
  int m = lane & 15;
  int kb = (lane >> 4) << 3;
  const _Float16* p = base + (size_t)(m0 + m) * ld + k0 + kb;
  v8h lo = *(const v8h*)(p);
  v8h hi = *(const v8h*)(p + 16);
  v16h r;
#pragma unroll
  for (int j = 0; j < 8; ++j) { r[j] = lo[j]; r[j + 8] = hi[j]; }
  return r;
}

// B-fragment: 32x16 f16, element [k][n], source stored k-contiguous per
// column (row n of baseT holds k contiguously).
__device__ __forceinline__ v16h load_b_frag_t(const _Float16* baseT, int ld,
                                              int k0, int n0) {
  int lane = threadIdx.x & 31;
  int n = lane & 15;
  int kb = (lane >> 4) << 3;
  const _Float16* p = baseT + (size_t)(n0 + n) * ld + k0 + kb;
  v8h lo = *(const v8h*)(p);
  v8h hi = *(const v8h*)(p + 16);
  v16h r;
#pragma unroll
  for (int j = 0; j < 8; ++j) { r[j] = lo[j]; r[j + 8] = hi[j]; }
  return r;
}

// ---------------------------------------------------------------------------
// CDNA5 async global->LDS copy (per-lane 16B), tracked by ASYNCcnt.
// ---------------------------------------------------------------------------
__device__ __forceinline__ void async_copy_b128(const void* gptr,
                                                unsigned lds_off) {
  asm volatile("global_load_async_to_lds_b128 %0, %1, off"
               :: "v"(lds_off), "v"(gptr) : "memory");
}
__device__ __forceinline__ void wait_async0() {
  asm volatile("s_wait_asynccnt 0" ::: "memory");
}

// ---------------------------------------------------------------------------
// Kernel 1: convert x (f32) -> xh (f16), row-major [MROWS][DMODEL]
// ---------------------------------------------------------------------------
__global__ void cvt_x_kernel(const float* __restrict__ x,
                             _Float16* __restrict__ xh, int n) {
  int i = blockIdx.x * blockDim.x + threadIdx.x;
  if (i < n) xh[i] = (_Float16)x[i];
}

// ---------------------------------------------------------------------------
// Kernel 2: transpose + convert weights: W f32 [K][N] -> WT f16 [N][K]
// ---------------------------------------------------------------------------
__global__ void transpose_w_kernel(const float* __restrict__ Wq,
                                   const float* __restrict__ Wk,
                                   const float* __restrict__ Wv,
                                   const float* __restrict__ Wo,
                                   _Float16* __restrict__ WT) {
  __shared__ float tile[16][17];
  const float* W = (blockIdx.z == 0) ? Wq
                 : (blockIdx.z == 1) ? Wk
                 : (blockIdx.z == 2) ? Wv : Wo;
  _Float16* dst = WT + (size_t)blockIdx.z * DMODEL * DMODEL;
  int k0 = blockIdx.x * 16, n0 = blockIdx.y * 16;
  int tx = threadIdx.x, ty = threadIdx.y;
  tile[ty][tx] = W[(size_t)(k0 + ty) * DMODEL + (n0 + tx)];
  __syncthreads();
  dst[(size_t)(n0 + ty) * DMODEL + (k0 + tx)] = (_Float16)tile[tx][ty];
}

// ---------------------------------------------------------------------------
// Kernel 3: QKV projection. Each wave: 32 rows x 64 cols (8 WMMA / k-step,
// B fragments reused across two A fragments). Scatter epilogue:
//   z=0: Qh [B,H,S,Hd] f16 (pre-scaled 0.125); z=1: Kh [B,H,S,Hd];
//   z=2: VT [B,H,Hd,S] f16.
// ---------------------------------------------------------------------------
__global__ void gemm_qkv_kernel(const _Float16* __restrict__ xh,
                                const _Float16* __restrict__ WT,
                                const float* __restrict__ bq,
                                const float* __restrict__ bk,
                                const float* __restrict__ bv,
                                _Float16* __restrict__ Qh,
                                _Float16* __restrict__ Kh,
                                _Float16* __restrict__ VT) {
  int lane = threadIdx.x & 31, wave = threadIdx.x >> 5;
  int z = blockIdx.z;
  int m0 = (blockIdx.x * 4 + wave) * 32;
  int n0 = blockIdx.y * 64;
  const _Float16* wt = WT + (size_t)z * DMODEL * DMODEL;
  v8f acc[2][4] = {};
  for (int k = 0; k < DMODEL; k += 32) {
    v16h a0 = load_a_frag(xh, DMODEL, m0, k);
    v16h a1 = load_a_frag(xh, DMODEL, m0 + 16, k);
#pragma unroll
    for (int t = 0; t < 4; ++t) {
      v16h bf = load_b_frag_t(wt, DMODEL, k, n0 + 16 * t);
      acc[0][t] = wmma_f16(a0, bf, acc[0][t]);
      acc[1][t] = wmma_f16(a1, bf, acc[1][t]);
    }
  }
  const float* bias = (z == 0) ? bq : (z == 1) ? bk : bv;
  int nlo = lane & 15;
  int mhi = (lane >> 4) << 3;
#pragma unroll
  for (int u = 0; u < 2; ++u)
#pragma unroll
    for (int t = 0; t < 4; ++t) {
      int n = n0 + 16 * t + nlo;
      float bn = bias[n];
      int h = n >> 6, hd = n & 63;
#pragma unroll
      for (int r = 0; r < 8; ++r) {
        int m = m0 + 16 * u + r + mhi;
        int s = m >> 1, b = m & 1;       // row = s*B + b (B==2)
        int bh = b * NHEAD + h;
        float v = acc[u][t][r] + bn;
        if (z == 0)
          Qh[((size_t)bh * S_LEN + s) * HDIM + hd] = (_Float16)(v * 0.125f);
        else if (z == 1)
          Kh[((size_t)bh * S_LEN + s) * HDIM + hd] = (_Float16)v;
        else
          VT[((size_t)bh * HDIM + hd) * S_LEN + s] = (_Float16)v;
      }
    }
}

// ---------------------------------------------------------------------------
// Kernel 4: causal flash attention with block-cooperative, double-buffered
// async LDS staging of K/V tiles. Block = 4 waves = one 64-query strip of
// one (b,h); wave w owns queries q0 = strip + 16w. Per 32-key tile:
// scores via WMMA from LDS K tile, online softmax, P round-trip through
// per-wave LDS (C-layout -> A-layout), P@V via WMMA from LDS V tile.
// ---------------------------------------------------------------------------
__global__ void flash_attn_kernel(const _Float16* __restrict__ Qh,
                                  const _Float16* __restrict__ Kh,
                                  const _Float16* __restrict__ VT,
                                  _Float16* __restrict__ Ah) {
  __shared__ __align__(32) _Float16 sK[2][32 * 64];  // [s][d]
  __shared__ __align__(32) _Float16 sV[2][64 * 32];  // [d][s]
  __shared__ __align__(32) _Float16 sP[4][16 * 32];
  int lane = threadIdx.x & 31, wave = threadIdx.x >> 5;
  int tid = threadIdx.x;
  int bh = blockIdx.y;                 // b*NHEAD + h
  int b = bh >> 4, h = bh & 15;
  int q0 = blockIdx.x * 64 + wave * 16;
  int smax = blockIdx.x * 64 + 63;     // block-uniform causal extent
  const _Float16* Qb = Qh + (size_t)bh * S_LEN * HDIM;
  const _Float16* Kb = Kh + (size_t)bh * S_LEN * HDIM;
  const _Float16* Vb = VT + (size_t)bh * HDIM * S_LEN;

  v16h aq0 = load_a_frag(Qb, HDIM, q0, 0);
  v16h aq1 = load_a_frag(Qb, HDIM, q0, 32);

  v8f acc[4] = {};
  float m_i[8], l_i[8];
#pragma unroll
  for (int r = 0; r < 8; ++r) { m_i[r] = -1e30f; l_i[r] = 0.f; }
  int nlo = lane & 15, mhi = (lane >> 4) << 3;
  _Float16* myP = sP[wave];

  // Cooperative async stage of one 32-key K tile (4KB) + V tile (4KB).
  auto stage = [&](int buf, int s0) {
    // K rows s0..s0+31 are 4KB contiguous in [S][Hd] f16.
    const char* gk = (const char*)(Kb + (size_t)s0 * HDIM);
    unsigned lk = (unsigned)(uintptr_t)(&sK[buf][0]);
#pragma unroll
    for (int p = 0; p < 2; ++p) {
      unsigned off = (unsigned)tid * 16u + (unsigned)p * 2048u;
      async_copy_b128(gk + off, lk + off);
    }
    // V: 64 rows of 64B (stride S_LEN halves).
    unsigned lv = (unsigned)(uintptr_t)(&sV[buf][0]);
#pragma unroll
    for (int p = 0; p < 2; ++p) {
      int c = tid + p * 128;           // 0..255 16B-chunks
      int d = c >> 2, part = c & 3;
      const char* gv =
          (const char*)(Vb + (size_t)d * S_LEN + s0) + part * 16;
      async_copy_b128(gv, lv + (unsigned)(d * 64 + part * 16));
    }
  };

  stage(0, 0);
  int buf = 0;
  for (int s0 = 0; s0 <= smax; s0 += 32) {
    wait_async0();          // my staged copies for sK/sV[buf] are done
    __syncthreads();        // everyone's are; prior compute all retired
    if (s0 + 32 <= smax) stage(buf ^ 1, s0 + 32);  // overlap next tile

    if (s0 <= q0 + 15) {
      const _Float16* tK = sK[buf];
      const _Float16* tV = sV[buf];
      // ---- scores 16x32, K-dim = Hd (64)
      v8f sc[2];
#pragma unroll
      for (int f = 0; f < 2; ++f) {
        v8f c = {};
        c = wmma_f16(aq0, load_b_frag_t(tK, HDIM, 0,  16 * f), c);
        c = wmma_f16(aq1, load_b_frag_t(tK, HDIM, 32, 16 * f), c);
        sc[f] = c;
      }
      // ---- causal mask
      if (s0 + 31 > q0) {
#pragma unroll
        for (int f = 0; f < 2; ++f)
#pragma unroll
          for (int r = 0; r < 8; ++r) {
            int q = q0 + r + mhi, s = s0 + 16 * f + nlo;
            if (s > q) sc[f][r] = -1e30f;
          }
      }
      // ---- online softmax row stats
      float fac[8];
#pragma unroll
      for (int r = 0; r < 8; ++r) {
        float mx = fmaxf(sc[0][r], sc[1][r]);
#pragma unroll
        for (int off = 8; off >= 1; off >>= 1)
          mx = fmaxf(mx, __shfl_xor(mx, off, 32));
        float nm = fmaxf(m_i[r], mx);
        fac[r] = __expf(m_i[r] - nm);
        float p0 = __expf(sc[0][r] - nm);
        float p1 = __expf(sc[1][r] - nm);
        sc[0][r] = p0; sc[1][r] = p1;
        float rs = p0 + p1;
#pragma unroll
        for (int off = 8; off >= 1; off >>= 1)
          rs += __shfl_xor(rs, off, 32);
        l_i[r] = l_i[r] * fac[r] + rs;
        m_i[r] = nm;
      }
#pragma unroll
      for (int t = 0; t < 4; ++t)
#pragma unroll
        for (int r = 0; r < 8; ++r)
          acc[t][r] *= fac[r];
      // ---- P: C-layout -> per-wave LDS row-major f16 [16][32]
#pragma unroll
      for (int r = 0; r < 8; ++r) {
        int m = r + mhi;
        myP[m * 32 + nlo]      = (_Float16)sc[0][r];
        myP[m * 32 + nlo + 16] = (_Float16)sc[1][r];
      }
      asm volatile("s_wait_dscnt 0" ::: "memory");
      // ---- reload P as A-fragment
      v16h ap;
      {
        int mm = lane & 15;
        int kb = (lane >> 4) << 3;
        const _Float16* p = myP + mm * 32 + kb;
        v8h lo = *(const v8h*)(p);
        v8h hi = *(const v8h*)(p + 16);
#pragma unroll
        for (int j = 0; j < 8; ++j) { ap[j] = lo[j]; ap[j + 8] = hi[j]; }
      }
      // ---- acc += P @ V from LDS V tile [d][s], ld = 32
#pragma unroll
      for (int t = 0; t < 4; ++t)
        acc[t] = wmma_f16(ap, load_b_frag_t(tV, 32, 0, 16 * t), acc[t]);
    }
    buf ^= 1;
  }
  // ---- epilogue: normalize, scatter to Ah [S,B,D] f16
#pragma unroll
  for (int t = 0; t < 4; ++t)
#pragma unroll
    for (int r = 0; r < 8; ++r) {
      int q = q0 + r + mhi;
      int col = h * HDIM + 16 * t + nlo;
      float v = acc[t][r] / l_i[r];
      Ah[((size_t)q * BATCH + b) * DMODEL + col] = (_Float16)v;
    }
}

// ---------------------------------------------------------------------------
// Kernel 5: output projection: d_out = Ah @ Wo + bo (f32 out), 32x64/wave
// ---------------------------------------------------------------------------
__global__ void gemm_out_kernel(const _Float16* __restrict__ Ah,
                                const _Float16* __restrict__ WoT,
                                const float* __restrict__ bo,
                                float* __restrict__ out) {
  int lane = threadIdx.x & 31, wave = threadIdx.x >> 5;
  int m0 = (blockIdx.x * 4 + wave) * 32;
  int n0 = blockIdx.y * 64;
  v8f acc[2][4] = {};
  for (int k = 0; k < DMODEL; k += 32) {
    v16h a0 = load_a_frag(Ah, DMODEL, m0, k);
    v16h a1 = load_a_frag(Ah, DMODEL, m0 + 16, k);
#pragma unroll
    for (int t = 0; t < 4; ++t) {
      v16h bf = load_b_frag_t(WoT, DMODEL, k, n0 + 16 * t);
      acc[0][t] = wmma_f16(a0, bf, acc[0][t]);
      acc[1][t] = wmma_f16(a1, bf, acc[1][t]);
    }
  }
  int nlo = lane & 15, mhi = (lane >> 4) << 3;
#pragma unroll
  for (int u = 0; u < 2; ++u)
#pragma unroll
    for (int t = 0; t < 4; ++t) {
      int n = n0 + 16 * t + nlo;
      float bn = bo[n];
#pragma unroll
      for (int r = 0; r < 8; ++r) {
        int m = m0 + 16 * u + r + mhi;
        out[(size_t)m * DMODEL + n] = acc[u][t][r] + bn;
      }
    }
}

// ---------------------------------------------------------------------------
// Launch. Inputs: x, mask, Wq, bq, Wk, bk, Wv, bv, Wo, bo.
// Workspace (f16 elements, 48 MB): xh | WT(x4) | Qh | Kh | VT | Ah
// ---------------------------------------------------------------------------
extern "C" void kernel_launch(void* const* d_in, const int* in_sizes, int n_in,
                              void* d_out, int out_size, void* d_ws,
                              size_t ws_size, hipStream_t stream) {
  const float* x  = (const float*)d_in[0];
  const float* Wq = (const float*)d_in[2];
  const float* bq = (const float*)d_in[3];
  const float* Wk = (const float*)d_in[4];
  const float* bk = (const float*)d_in[5];
  const float* Wv = (const float*)d_in[6];
  const float* bv = (const float*)d_in[7];
  const float* Wo = (const float*)d_in[8];
  const float* bo = (const float*)d_in[9];
  float* out = (float*)d_out;

  const size_t NWORDS = (size_t)MROWS * DMODEL;      // 4,194,304
  _Float16* xh = (_Float16*)d_ws;
  _Float16* WT = xh + NWORDS;
  _Float16* Qh = WT + 4 * (size_t)DMODEL * DMODEL;
  _Float16* Kh = Qh + NWORDS;
  _Float16* VT = Kh + NWORDS;
  _Float16* Ah = VT + NWORDS;

  cvt_x_kernel<<<(int)(NWORDS / 256), 256, 0, stream>>>(x, xh, (int)NWORDS);

  transpose_w_kernel<<<dim3(DMODEL / 16, DMODEL / 16, 4), dim3(16, 16), 0,
                       stream>>>(Wq, Wk, Wv, Wo, WT);

  gemm_qkv_kernel<<<dim3(MROWS / 128, DMODEL / 64, 3), 128, 0, stream>>>(
      xh, WT, bq, bk, bv, Qh, Kh, VT);

  flash_attn_kernel<<<dim3(S_LEN / 64, BATCH * NHEAD), 128, 0, stream>>>(
      Qh, Kh, VT, Ah);

  gemm_out_kernel<<<dim3(MROWS / 128, DMODEL / 64), 128, 0, stream>>>(
      Ah, WT + 3 * (size_t)DMODEL * DMODEL, bo, out);
}
